// FrankensteinsRNN_61211873903133
// MI455X (gfx1250) — matrix operand order repaired and checked by reading, not compiled, not run on hardware
//
#include <hip/hip_runtime.h>
#include <hip/hip_bf16.h>
#include <math.h>

// ---------------------------------------------------------------------------
// FrankensteinsRNN on MI455X (gfx1250, wave32, WMMA).
//
//   1) gather+cast:      xs_f16  = f16(emb[tokens])                [S,512]
//   2) transpose+cast:   WT[n,k] = f16(W[k,n])  for Wx1,Wh1,Wx2,Wh2,Wo
//                        (Wo padded to N=50304 = 393*128, zeros)
//   3) WMMA GEMM:        A1 = xs_f16 @ Wx1 + b1                    fp32
//   4) WMMA recurrence:  h1[t]=tanh(A1[t]+h1[t-1]@Wh1) -> H1_f16
//                        (Wh^T K<256 LDS-resident, K>=256 streamed from L2)
//   5) WMMA GEMM:        A2 = H1_f16 @ Wx2 + b2                    fp32
//   6) WMMA recurrence:  h2[t]=tanh(A2[t]+h2[t-1]@Wh2) -> H2_f16
//   7) WMMA GEMM:        Out = H2_f16 @ Wo + bo -> d_out, NT stores
// ---------------------------------------------------------------------------

#define S_LEN 2048
#define H_DIM 512
#define V_DIM 50257
#define V_PAD 50304            // 393 * 128
#define K_LDS 256              // K-half of Wh^T kept LDS-resident (256 KB)
#define RNN_LDS_BYTES (H_DIM * K_LDS * 2 + H_DIM * 2)   // 256 KB + 1 KB

typedef _Float16 v16h __attribute__((ext_vector_type(16)));
typedef _Float16 v8h  __attribute__((ext_vector_type(8)));
typedef float    v8f  __attribute__((ext_vector_type(8)));

// ---- WMMA fragment helpers -------------------------------------------------
// 16-bit A matrix 16x32: lanes 0-15 hold K={0..15},
// lanes 16-31 hold K={16..31} (contiguous, matches B); rows = lane&15.
__device__ __forceinline__ v16h load_a_frag(const _Float16* __restrict__ rowPtr, int g) {
  v8h lo = *(const v8h*)(rowPtr + g * 16);
  v8h hi = *(const v8h*)(rowPtr + g * 16 + 8);
  v16h a;
#pragma unroll
  for (int i = 0; i < 8; ++i) { a[i] = lo[i]; a[i + 8] = hi[i]; }
  return a;
}

// B matrix 32x16 with B stored transposed [N,K]: one contiguous 32B load.
__device__ __forceinline__ v16h load_b_frag(const _Float16* __restrict__ colRowPtr, int g) {
  return *(const v16h*)(colRowPtr + g * 16);
}

__device__ __forceinline__ v8f wmma_f16(v16h a, v16h b, v8f c) {
  return __builtin_amdgcn_wmma_f32_16x16x32_f16(false, a, false, b, (short)0, c,
                                                false, false);
}

// Branchless tanh on the hardware transcendental pipe:
// tanh(x) = 1 - 2/(e^{2x}+1);  e^{2x} = 2^{x * 2*log2(e)}.
// v_exp_f32 + v_rcp_f32, ~1ulp each; saturates correctly at +-inf.
__device__ __forceinline__ float fast_tanh(float x) {
  float e = __builtin_amdgcn_exp2f(x * 2.8853900817779268f);
  return 1.0f - 2.0f * __builtin_amdgcn_rcpf(e + 1.0f);
}

// ---- 1) embedding gather + cast to f16 ------------------------------------
__global__ __launch_bounds__(256) void k_gather_cast(const int* __restrict__ tokens,
                                                     const float* __restrict__ emb,
                                                     _Float16* __restrict__ xs16) {
  const int t   = blockIdx.x;
  const int tok = tokens[t];
  const float* src = emb + (size_t)tok * H_DIM;
  _Float16*    dst = xs16 + (size_t)t * H_DIM;
  const int i = threadIdx.x;
  dst[i]       = (_Float16)src[i];
  dst[i + 256] = (_Float16)src[i + 256];
}

// ---- 2) transpose [K,N] fp32 -> [Npad,K] f16 (zero padded) -----------------
__global__ __launch_bounds__(256) void k_transpose_cast(const float* __restrict__ W,
                                                        _Float16* __restrict__ WT,
                                                        int N, int K) {
  __shared__ float tile[16][17];
  const int n = blockIdx.x * 16 + threadIdx.x;
  const int k = blockIdx.y * 16 + threadIdx.y;
  tile[threadIdx.y][threadIdx.x] = (n < N) ? W[(size_t)k * N + n] : 0.0f;
  __syncthreads();
  const int wn = blockIdx.x * 16 + threadIdx.y;
  const int wk = blockIdx.y * 16 + threadIdx.x;
  WT[(size_t)wn * K + wk] = (_Float16)tile[threadIdx.x][threadIdx.y];
}

// ---- 3/5/7) C[M,Nstore] = A_f16[M,K] @ BT_f16[Npad,K] + bias ---------------
// 256 threads = 8 waves; block tile 128x128; wave tile 32x64 (2 M x 4 N frags).
// NT=true -> non-temporal stores (streaming 411MB output; keep Wo/H2 in L2).
template <bool NT>
__global__ __launch_bounds__(256) void k_gemm_f16(const _Float16* __restrict__ A,
                                                  const _Float16* __restrict__ BT,
                                                  const float* __restrict__ bias,
                                                  float* __restrict__ C,
                                                  int K, int Nstore, int ldC) {
  const int tid  = threadIdx.x;
  const int lane = tid & 31;
  const int wid  = tid >> 5;
  const int wm   = wid & 3;
  const int wn   = wid >> 2;
  const int ln   = lane & 15;
  const int g    = lane >> 4;

  const int mBase = blockIdx.y * 128 + wm * 32;
  const int nBase = blockIdx.x * 128 + wn * 64;

  v8f acc[2][4] = {};
  const _Float16* aRow0 = A + (size_t)(mBase + ln) * K;
  const _Float16* aRow1 = A + (size_t)(mBase + 16 + ln) * K;
  const _Float16* bRow[4];
#pragma unroll
  for (int nf = 0; nf < 4; ++nf)
    bRow[nf] = BT + (size_t)(nBase + nf * 16 + ln) * K;

  for (int kb = 0; kb < K; kb += 32) {
    v16h a0 = load_a_frag(aRow0 + kb, g);
    v16h a1 = load_a_frag(aRow1 + kb, g);
#pragma unroll
    for (int nf = 0; nf < 4; ++nf) {
      v16h b = load_b_frag(bRow[nf] + kb, g);
      acc[0][nf] = wmma_f16(a0, b, acc[0][nf]);
      acc[1][nf] = wmma_f16(a1, b, acc[1][nf]);
    }
  }

  // C/D layout: VGPR r -> M = base + r + g*8, N = base + (lane&15)
#pragma unroll
  for (int nf = 0; nf < 4; ++nf) {
    const int n = nBase + nf * 16 + ln;
    if (n < Nstore) {
      const float bv = bias[n];
#pragma unroll
      for (int mf = 0; mf < 2; ++mf) {
#pragma unroll
        for (int r = 0; r < 8; ++r) {
          const int m = mBase + mf * 16 + r + g * 8;
          float v = acc[mf][nf][r] + bv;
          float* p = C + (size_t)m * ldC + n;
          if (NT) __builtin_nontemporal_store(v, p);
          else    *p = v;
        }
      }
    }
  }
}

// ---- 4/6) sequential recurrence: h[t] = tanh(Apre[t] + h[t-1] @ Wh) --------
// One workgroup, 512 threads = 16 waves; wave w owns output cols [32w,32w+32).
// Wh^T rows (K < K_LDS) live in LDS; (K >= K_LDS) streamed from L2 each step,
// interleaved per K-tile so DS and VMEM pipes overlap under the WMMA chain.
__global__ __launch_bounds__(512) void k_rnn_layer(const float* __restrict__ Apre,
                                                   const _Float16* __restrict__ WhT,
                                                   _Float16* __restrict__ Hout) {
  extern __shared__ char smem[];
  _Float16* whL = (_Float16*)smem;                         // [H_DIM][K_LDS]
  _Float16* h16 = (_Float16*)(smem + H_DIM * K_LDS * 2);   // [H_DIM]

  const int tid  = threadIdx.x;
  const int lane = tid & 31;
  const int w    = tid >> 5;
  const int ln   = lane & 15;
  const int g    = lane >> 4;

  // Preload Wh^T K<K_LDS slice: thread tid copies row tid (256 halves).
  {
    const _Float16* src = WhT + (size_t)tid * H_DIM;
    _Float16* dst = whL + tid * K_LDS;
#pragma unroll
    for (int k = 0; k < K_LDS; k += 8)
      *(v8h*)(dst + k) = *(const v8h*)(src + k);
  }
  h16[tid] = (_Float16)0.0f;
  __syncthreads();

  const int n0 = w * 32 + ln;
  const int n1 = n0 + 16;
  const _Float16* g0p = WhT + (size_t)n0 * H_DIM + K_LDS;  // global half, col n0
  const _Float16* g1p = WhT + (size_t)n1 * H_DIM + K_LDS;  // global half, col n1
  const _Float16* l0p = whL + n0 * K_LDS;                  // LDS half, col n0
  const _Float16* l1p = whL + n1 * K_LDS;                  // LDS half, col n1

  for (int t = 0; t < S_LEN; ++t) {
    // Hoisted: issues right after the previous barrier, latency hidden
    // under the WMMA chain (not needed until the epilogue).
    const float ap0 = Apre[t * H_DIM + n0];
    const float ap1 = Apre[t * H_DIM + n1];

    v8f acc0 = {};
    v8f acc1 = {};
#pragma unroll 2
    for (int kt = 0; kt < 8; ++kt) {
      const int kb = kt * 32;
      // K-tile from LDS-resident half (K = kb)
      v16h aL = load_a_frag(&h16[kb], g);
      v16h bL0 = *(const v16h*)(l0p + kb + g * 16);
      v16h bL1 = *(const v16h*)(l1p + kb + g * 16);
      acc0 = wmma_f16(aL, bL0, acc0);
      acc1 = wmma_f16(aL, bL1, acc1);
      // K-tile from global/L2 half (K = K_LDS + kb)
      v16h aG = load_a_frag(&h16[K_LDS + kb], g);
      v16h bG0 = load_b_frag(g0p + kb, g);
      v16h bG1 = load_b_frag(g1p + kb, g);
      acc0 = wmma_f16(aG, bG0, acc0);
      acc1 = wmma_f16(aG, bG1, acc1);
    }
    __syncthreads();  // everyone done reading h[t-1]
    if (lane < 16) {  // row 0 of the replicated tile: element 0, lanes 0..15
      const float v0 = fast_tanh(acc0[0] + ap0);
      const float v1 = fast_tanh(acc1[0] + ap1);
      const _Float16 f0 = (_Float16)v0;
      const _Float16 f1 = (_Float16)v1;
      h16[n0] = f0;
      h16[n1] = f1;
      Hout[(size_t)t * H_DIM + n0] = f0;
      Hout[(size_t)t * H_DIM + n1] = f1;
    }
    __syncthreads();  // h[t] visible to all waves
  }
}

// ---------------------------------------------------------------------------
extern "C" void kernel_launch(void* const* d_in, const int* in_sizes, int n_in,
                              void* d_out, int out_size, void* d_ws, size_t ws_size,
                              hipStream_t stream) {
  (void)in_sizes; (void)n_in; (void)out_size; (void)ws_size;

  const int*   x    = (const int*)d_in[0];
  const float* emb  = (const float*)d_in[1];
  const float* Wx1  = (const float*)d_in[2];
  const float* Wh1  = (const float*)d_in[3];
  const float* b1   = (const float*)d_in[4];
  const float* Wx2  = (const float*)d_in[5];
  const float* Wh2  = (const float*)d_in[6];
  const float* b2   = (const float*)d_in[7];
  const float* Wo   = (const float*)d_in[8];
  const float* bo   = (const float*)d_in[9];
  float* out = (float*)d_out;

  // Workspace carve-up (256B aligned). Total ~68.3 MB.
  char* ws = (char*)d_ws;
  size_t off = 0;
  auto carve = [&](size_t bytes) -> char* {
    char* p = ws + off;
    off = (off + bytes + 255) & ~(size_t)255;
    return p;
  };
  _Float16* xs16  = (_Float16*)carve((size_t)S_LEN * H_DIM * 2);
  _Float16* Wx1T  = (_Float16*)carve((size_t)H_DIM * H_DIM * 2);
  _Float16* Wh1T  = (_Float16*)carve((size_t)H_DIM * H_DIM * 2);
  _Float16* Wx2T  = (_Float16*)carve((size_t)H_DIM * H_DIM * 2);
  _Float16* Wh2T  = (_Float16*)carve((size_t)H_DIM * H_DIM * 2);
  _Float16* WoT   = (_Float16*)carve((size_t)V_PAD * H_DIM * 2);
  float*    A1    = (float*)carve((size_t)S_LEN * H_DIM * 4);
  float*    A2    = (float*)carve((size_t)S_LEN * H_DIM * 4);
  _Float16* H1_16 = (_Float16*)carve((size_t)S_LEN * H_DIM * 2);
  _Float16* H2_16 = (_Float16*)carve((size_t)S_LEN * H_DIM * 2);

  // 1) gather embeddings -> f16
  k_gather_cast<<<S_LEN, 256, 0, stream>>>(x, emb, xs16);

  // 2) transpose+cast weights to B^T layout
  {
    dim3 blk(16, 16);
    dim3 grd512(H_DIM / 16, H_DIM / 16);
    k_transpose_cast<<<grd512, blk, 0, stream>>>(Wx1, Wx1T, H_DIM, H_DIM);
    k_transpose_cast<<<grd512, blk, 0, stream>>>(Wh1, Wh1T, H_DIM, H_DIM);
    k_transpose_cast<<<grd512, blk, 0, stream>>>(Wx2, Wx2T, H_DIM, H_DIM);
    k_transpose_cast<<<grd512, blk, 0, stream>>>(Wh2, Wh2T, H_DIM, H_DIM);
    dim3 grdWo(V_PAD / 16, H_DIM / 16);
    k_transpose_cast<<<grdWo, blk, 0, stream>>>(Wo, WoT, V_DIM, H_DIM);
  }

  // 3) A1 = xs @ Wx1 + b1 (regular stores: consumed from L2 by recurrence)
  k_gemm_f16<false><<<dim3(H_DIM / 128, S_LEN / 128), 256, 0, stream>>>(
      xs16, Wx1T, b1, A1, H_DIM, H_DIM, H_DIM);

  // 4) layer-1 recurrence (sequential, LDS-resident K-half)
  k_rnn_layer<<<1, 512, RNN_LDS_BYTES, stream>>>(A1, Wh1T, H1_16);

  // 5) A2 = H1 @ Wx2 + b2
  k_gemm_f16<false><<<dim3(H_DIM / 128, S_LEN / 128), 256, 0, stream>>>(
      H1_16, Wx2T, b2, A2, H_DIM, H_DIM, H_DIM);

  // 6) layer-2 recurrence
  k_rnn_layer<<<1, 512, RNN_LDS_BYTES, stream>>>(A2, Wh2T, H2_16);

  // 7) Out = H2 @ Wo + bo; NT stores stream the 411MB result past L2
  k_gemm_f16<true><<<dim3(V_PAD / 128, S_LEN / 128), 256, 0, stream>>>(
      H2_16, WoT, bo, out, H_DIM, V_DIM, V_DIM);
}
